// Attn_head_37589553774673
// MI455X (gfx1250) — compile-verified, matrix operand order checked by reading, not run on hardware
//
#include <hip/hip_runtime.h>
#include <hip/hip_bf16.h>

// ---------------- problem constants (fixed by setup_inputs) ----------------
constexpr int BATCH = 2;
constexpr int SEQ   = 2048;
constexpr int HID   = 1024;
constexpr int NH    = 16;
constexpr int HD    = 64;
constexpr float NEGM = -1000000.0f;

// ---------------- WMMA plumbing ----------------
typedef __attribute__((ext_vector_type(16))) __bf16 bf16x16;
typedef __attribute__((ext_vector_type(8)))  float  floatx8;
typedef __attribute__((ext_vector_type(4)))  unsigned uintx4;
typedef __attribute__((ext_vector_type(8)))  int      intx8;
typedef __attribute__((ext_vector_type(4)))  int      intx4;

union AFrag { bf16x16 v; unsigned u[8]; };

__device__ __forceinline__ floatx8 wmma_bf16(bf16x16 a, bf16x16 b, floatx8 c) {
  return __builtin_amdgcn_wmma_f32_16x16x32_bf16(
      false, a, false, b, (short)0, c, false, false);
}

__device__ __forceinline__ unsigned short f2bf(float f) {
  union { float f; unsigned u; } a; a.f = f;
  unsigned r = a.u + 0x7FFFu + ((a.u >> 16) & 1u);   // round-to-nearest-even
  return (unsigned short)(r >> 16);
}
__device__ __forceinline__ unsigned packbf(float x, float y) {
  return (unsigned)f2bf(x) | ((unsigned)f2bf(y) << 16);
}
// K-pair base for 16-bit A/B fragments: lane half g, dword v
__device__ __forceinline__ int kpat(int v, int g) {
  return ((v & 3) << 1) + (g << 3) + ((v >> 2) << 4);
}

// LDS byte offset of a __shared__ pointer (generic -> addrspace(3))
__device__ __forceinline__ unsigned lds_off(const void* p) {
  return (unsigned)(size_t)(__attribute__((address_space(3))) const void*)p;
}

// ---- Tensor Data Mover: 2D bf16 tile Global -> LDS (6-arg clang-23 builtin) ----
// pad codes: interval 2^(ic+1) dwords, amount (ac+1) dwords
__device__ __forceinline__ void tdm_load_2d_bf16(
    unsigned ldsOff, const void* gptr,
    unsigned tile_w, unsigned tile_h, unsigned row_stride_elems,
    unsigned pad_ic, unsigned pad_ac, bool pad_en)
{
  unsigned long long ga = (unsigned long long)(size_t)gptr;
  uintx4 g0;
  g0.x = 1u;                                   // count=1, user descriptor
  g0.y = ldsOff;                               // LDS byte address
  g0.z = (unsigned)(ga & 0xFFFFFFFFu);         // global addr [31:0]
  g0.w = (unsigned)((ga >> 32) & 0x01FFFFFFu) | (2u << 30);  // [56:32] | type=2
  unsigned w0 = (1u << 16);                    // data_size = 2 bytes
  if (pad_en) w0 |= (1u << 20) | (pad_ic << 22) | (pad_ac << 25);
  intx8 g1;
  g1[0] = (int)w0;
  g1[1] = (int)((tile_w & 0xFFFFu) << 16);                                // tensor_dim0 lo
  g1[2] = (int)(((tile_w >> 16) & 0xFFFFu) | ((tile_h & 0xFFFFu) << 16)); // dim0 hi | dim1 lo
  g1[3] = (int)(((tile_h >> 16) & 0xFFFFu) | ((tile_w & 0xFFFFu) << 16)); // dim1 hi | tile_dim0
  g1[4] = (int)(tile_h & 0xFFFFu);                                        // tile_dim1
  g1[5] = (int)row_stride_elems;                                          // dim0_stride lo32
  g1[6] = 0;
  g1[7] = 0;
  intx4 z4 = {0, 0, 0, 0};
  intx8 z8 = {0, 0, 0, 0, 0, 0, 0, 0};
  __builtin_amdgcn_tensor_load_to_lds(g0, g1, z4, z4, z8, 0);
}

// =================== Kernel 1: QKV projection GEMM (bf16 WMMA) ===============
constexpr int MB = 128, NB = 64, KB = 32;
constexpr int LDA = 34, LDB = 34;   // padded bf16 strides (17-word rows)

__global__ __launch_bounds__(256) void qkv_gemm(
    const float* __restrict__ X,
    const float* __restrict__ W0, const float* __restrict__ W1, const float* __restrict__ W2,
    const float* __restrict__ B0, const float* __restrict__ B1, const float* __restrict__ B2,
    float* __restrict__ OF0, float* __restrict__ OF1, float* __restrict__ OF2,
    unsigned short* __restrict__ OB0, unsigned short* __restrict__ OB1,
    unsigned short* __restrict__ OB2)
{
  const int z = blockIdx.z;
  const float* W    = (z == 0) ? W0 : (z == 1) ? W1 : W2;
  const float* bias = (z == 0) ? B0 : (z == 1) ? B1 : B2;
  float* OutF       = (z == 0) ? OF0 : (z == 1) ? OF1 : OF2;
  unsigned short* OutB = (z == 0) ? OB0 : (z == 1) ? OB1 : OB2;

  const int n0 = blockIdx.x * NB;
  const int m0 = blockIdx.y * MB;

  __shared__ __align__(16) unsigned short sA[MB * LDA];   // [row][k]
  __shared__ __align__(16) unsigned short sB[NB * LDB];   // [n][k] (transposed)

  const int tid  = threadIdx.x;
  const int lane = tid & 31;
  const int wave = tid >> 5;
  const int g    = lane >> 4;
  const int ln   = lane & 15;
  const int wm   = wave >> 1;
  const int wn   = wave & 1;

  floatx8 acc[2][2] = {};

  for (int k0 = 0; k0 < HID; k0 += KB) {
    __syncthreads();
#pragma unroll
    for (int i = 0; i < 4; ++i) {                 // A: 128x32 fp32->bf16
      int qid = tid + i * 256;
      int row = qid >> 3;
      int kq  = (qid & 7) << 2;
      const float4 f = *(const float4*)(X + (size_t)(m0 + row) * HID + k0 + kq);
      unsigned* p = (unsigned*)(sA + row * LDA + kq);
      p[0] = packbf(f.x, f.y);
      p[1] = packbf(f.z, f.w);
    }
#pragma unroll
    for (int i = 0; i < 2; ++i) {                 // B transposed: 32x64
      int qid = tid + i * 256;
      int kk  = qid >> 4;
      int nq  = (qid & 15) << 2;
      const float4 f = *(const float4*)(W + (size_t)(k0 + kk) * HID + n0 + nq);
      sB[(nq + 0) * LDB + kk] = f2bf(f.x);
      sB[(nq + 1) * LDB + kk] = f2bf(f.y);
      sB[(nq + 2) * LDB + kk] = f2bf(f.z);
      sB[(nq + 3) * LDB + kk] = f2bf(f.w);
    }
    __syncthreads();

    AFrag a[2], bt[2];
#pragma unroll
    for (int mi = 0; mi < 2; ++mi) {
      const int row = wm * 32 + mi * 16 + ln;
#pragma unroll
      for (int v = 0; v < 8; ++v)
        a[mi].u[v] = *(const unsigned*)(sA + row * LDA + kpat(v, g));
    }
#pragma unroll
    for (int ni = 0; ni < 2; ++ni) {
      const int col = wn * 32 + ni * 16 + ln;
#pragma unroll
      for (int v = 0; v < 8; ++v)
        bt[ni].u[v] = *(const unsigned*)(sB + col * LDB + kpat(v, g));
    }
#pragma unroll
    for (int mi = 0; mi < 2; ++mi)
#pragma unroll
      for (int ni = 0; ni < 2; ++ni)
        acc[mi][ni] = wmma_bf16(a[mi].v, bt[ni].v, acc[mi][ni]);
  }

  // epilogue: + bias; values once, then straight-line store loops
  float val[2][2][8];
#pragma unroll
  for (int mi = 0; mi < 2; ++mi)
#pragma unroll
    for (int ni = 0; ni < 2; ++ni) {
      const float bv = bias[n0 + wn * 32 + ni * 16 + ln];
#pragma unroll
      for (int v = 0; v < 8; ++v)
        val[mi][ni][v] = acc[mi][ni][v] + bv;
    }

  if (OutF != nullptr) {                       // uniform branch, hoisted
#pragma unroll
    for (int mi = 0; mi < 2; ++mi)
#pragma unroll
      for (int ni = 0; ni < 2; ++ni) {
        const int colg = n0 + wn * 32 + ni * 16 + ln;
#pragma unroll
        for (int v = 0; v < 8; ++v) {
          const int rowg = m0 + wm * 32 + mi * 16 + v + g * 8;
          OutF[(size_t)rowg * HID + colg] = val[mi][ni][v];
        }
      }
  }
#pragma unroll
  for (int mi = 0; mi < 2; ++mi)
#pragma unroll
    for (int ni = 0; ni < 2; ++ni) {
      const int colg = n0 + wn * 32 + ni * 16 + ln;
#pragma unroll
      for (int v = 0; v < 8; ++v) {
        const int rowg = m0 + wm * 32 + mi * 16 + v + g * 8;
        OutB[(size_t)rowg * HID + colg] = f2bf(val[mi][ni][v]);
      }
    }
}

// ======= Kernel 2: flash attention (bf16 WMMA + double-buffered TDM) ========
constexpr int QB   = 128;  // q rows per workgroup (8 waves x 16)
constexpr int KBLK = 32;   // keys per step
constexpr int LDQ  = 72;   // 144B rows: 16B aligned, conflict-free
constexpr int LDK  = 72;   // produced by TDM pad (32dw interval + 4dw pad)
constexpr int LDV  = 34, LDP = 34;

__global__ __launch_bounds__(256) void attn_kernel(
    const unsigned short* __restrict__ Qb, const unsigned short* __restrict__ Kb,
    const unsigned short* __restrict__ Vb, const float* __restrict__ pad,
    float* __restrict__ Out)
{
  const int qb = blockIdx.x;     // 0..15
  const int h  = blockIdx.y;     // 0..15
  const int b  = blockIdx.z;     // 0..1
  const int q0 = qb * QB;

  __shared__ __align__(16) unsigned short sQ[QB * LDQ];        // [qrow 128][d 64]
  __shared__ __align__(16) unsigned short sK[2][KBLK * LDK];   // double-buffered (TDM)
  __shared__ __align__(16) unsigned short sV[HD * LDV];        // [d 64][key 32]
  __shared__ __align__(16) unsigned short sP[8][16 * LDP];     // per-wave P tiles

  const int tid  = threadIdx.x;
  const int lane = tid & 31;
  const int wave = tid >> 5;
  const int g    = lane >> 4;
  const int ln   = lane & 15;

  const int jmax = (q0 + QB) / KBLK;          // causal trip count (>= 4)
  const unsigned sK_off0 = lds_off(sK[0]);
  const unsigned sK_off1 = lds_off(sK[1]);

  // issue TDM for K tile 0 immediately — overlaps with Q staging
  if (wave == 0) {
    tdm_load_2d_bf16(sK_off0, Kb + (size_t)(b * SEQ) * HID + h * HD,
                     HD, KBLK, HID, /*32dw*/4, /*4dw*/3, true);
  }

  // ---- stage Q block once: 128x64 bf16 (b128 loads -> b128 LDS stores) ----
#pragma unroll
  for (int i = 0; i < 4; ++i) {
    int qid = tid + i * 256;
    int row = qid >> 3;
    int du  = (qid & 7) << 3;                 // 8 elems = 16B
    const uint4 f = *(const uint4*)(Qb + (size_t)(b * SEQ + q0 + row) * HID + h * HD + du);
    *(uint4*)(sQ + row * LDQ + du) = f;       // 144B row stride: 16B aligned
  }
  __syncthreads();

  AFrag aq[2];
#pragma unroll
  for (int c = 0; c < 2; ++c)
#pragma unroll
    for (int v = 0; v < 8; ++v)
      aq[c].u[v] = *(const unsigned*)(sQ + (wave * 16 + ln) * LDQ + c * 32 + kpat(v, g));

  float mrow[8], lrow[8];
#pragma unroll
  for (int v = 0; v < 8; ++v) { mrow[v] = -3.0e38f; lrow[v] = 0.0f; }
  floatx8 o[4] = {};

  for (int j = 0; j < jmax; ++j) {
    const int kb0 = j * KBLK;
    __syncthreads();   // prev iteration fully consumed sK[(j-1)&1] and sV

    // ---- issue TDM for NEXT K tile into the other buffer ----
    if (wave == 0 && (j + 1) < jmax) {
      tdm_load_2d_bf16(((j + 1) & 1) ? sK_off1 : sK_off0,
                       Kb + (size_t)(b * SEQ + kb0 + KBLK) * HID + h * HD,
                       HD, KBLK, HID, 4, 3, true);
    }
    // ---- V tile transposed [d][key], manual (TDM cannot transpose 16-bit) ----
    {
      int key = tid >> 3;                     // 0..31
      int du  = (tid & 7) << 3;               // 8 d-values
      const uint4 f = *(const uint4*)(Vb + (size_t)(b * SEQ + kb0 + key) * HID + h * HD + du);
      const unsigned w[4] = {f.x, f.y, f.z, f.w};
#pragma unroll
      for (int q = 0; q < 4; ++q) {
        sV[(du + 2 * q + 0) * LDV + key] = (unsigned short)(w[q] & 0xFFFFu);
        sV[(du + 2 * q + 1) * LDV + key] = (unsigned short)(w[q] >> 16);
      }
    }
    if (j + 1 < jmax) {                       // prefetch next V tile
      const char* nv = (const char*)(Vb + (size_t)(b * SEQ + kb0 + KBLK) * HID + h * HD);
      __builtin_prefetch(nv + tid * 16, 0, 1);
    }
    // tile j's TDM is complete when <=1 tensor op remains outstanding (in-order)
    if (wave == 0) {
      if (j + 1 < jmax) __builtin_amdgcn_s_wait_tensorcnt(1);
      else              __builtin_amdgcn_s_wait_tensorcnt(0);
    }
    __syncthreads();

    const unsigned short* sKj = sK[j & 1];

    // ---- S = Q * K^T (16 x 32) ----
    floatx8 s[2] = {};
#pragma unroll
    for (int f = 0; f < 2; ++f) {
      const int keyl = f * 16 + ln;
#pragma unroll
      for (int c = 0; c < 2; ++c) {
        AFrag bk;
#pragma unroll
        for (int v = 0; v < 8; ++v)
          bk.u[v] = *(const unsigned*)(sKj + keyl * LDK + c * 32 + kpat(v, g));
        s[f] = wmma_bf16(aq[c].v, bk.v, s[f]);
      }
    }

    // ---- scale + causal + pad mask (additive, like reference) ----
    const int qbase = q0 + wave * 16;
    const float pen0 = (1.0f - pad[b * SEQ + kb0 + ln])      * NEGM;
    const float pen1 = (1.0f - pad[b * SEQ + kb0 + 16 + ln]) * NEGM;
#pragma unroll
    for (int f = 0; f < 2; ++f) {
      const int   kj  = kb0 + f * 16 + ln;
      const float pen = f ? pen1 : pen0;
#pragma unroll
      for (int v = 0; v < 8; ++v) {
        const int qi = qbase + v + 8 * g;
        s[f][v] = s[f][v] * 0.125f + pen + ((kj > qi) ? NEGM : 0.0f);
      }
    }

    // ---- online softmax (row stats across 16-lane halves) ----
    float alpha[8];
#pragma unroll
    for (int v = 0; v < 8; ++v) {
      float mx = fmaxf(s[0][v], s[1][v]);
      mx = fmaxf(mx, __shfl_xor(mx, 1, 32));
      mx = fmaxf(mx, __shfl_xor(mx, 2, 32));
      mx = fmaxf(mx, __shfl_xor(mx, 4, 32));
      mx = fmaxf(mx, __shfl_xor(mx, 8, 32));
      const float mnew = fmaxf(mrow[v], mx);
      alpha[v] = __expf(mrow[v] - mnew);
      const float p0 = __expf(s[0][v] - mnew);
      const float p1 = __expf(s[1][v] - mnew);
      s[0][v] = p0; s[1][v] = p1;
      float rs = p0 + p1;
      rs += __shfl_xor(rs, 1, 32);
      rs += __shfl_xor(rs, 2, 32);
      rs += __shfl_xor(rs, 4, 32);
      rs += __shfl_xor(rs, 8, 32);
      lrow[v] = lrow[v] * alpha[v] + rs;
      mrow[v] = mnew;
    }

    // ---- P (C-layout) -> per-wave LDS -> A-layout fragment ----
    unsigned short* P = sP[wave];
#pragma unroll
    for (int f = 0; f < 2; ++f)
#pragma unroll
      for (int v = 0; v < 8; ++v)
        P[(v + 8 * g) * LDP + f * 16 + ln] = f2bf(s[f][v]);
    asm volatile("s_wait_dscnt 0x0" ::: "memory");

    AFrag ap;
#pragma unroll
    for (int v = 0; v < 8; ++v)
      ap.u[v] = *(const unsigned*)(P + ln * LDP + kpat(v, g));

    // ---- rescale O, then O += P * V ----
#pragma unroll
    for (int t = 0; t < 4; ++t)
#pragma unroll
      for (int v = 0; v < 8; ++v)
        o[t][v] *= alpha[v];
#pragma unroll
    for (int t = 0; t < 4; ++t) {
      AFrag bv;
      const int dcol = t * 16 + ln;
#pragma unroll
      for (int v = 0; v < 8; ++v)
        bv.u[v] = *(const unsigned*)(sV + dcol * LDV + kpat(v, g));
      o[t] = wmma_bf16(ap.v, bv.v, o[t]);
    }
  }

  // ---- normalize and store out[b, q, h*64 + d] ----
#pragma unroll
  for (int t = 0; t < 4; ++t) {
    const int d = t * 16 + ln;
#pragma unroll
    for (int v = 0; v < 8; ++v) {
      const int q = q0 + wave * 16 + v + 8 * g;
      Out[(size_t)(b * SEQ + q) * HID + h * HD + d] = o[t][v] / lrow[v];
    }
  }
}

// =================== host-side launch ==================
extern "C" void kernel_launch(void* const* d_in, const int* in_sizes, int n_in,
                              void* d_out, int out_size, void* d_ws, size_t ws_size,
                              hipStream_t stream) {
  const float* x   = (const float*)d_in[0];
  const float* pad = (const float*)d_in[1];
  const float* Wq  = (const float*)d_in[2];
  const float* bq  = (const float*)d_in[3];
  const float* Wk  = (const float*)d_in[4];
  const float* bk  = (const float*)d_in[5];
  const float* Wv  = (const float*)d_in[6];
  const float* bv  = (const float*)d_in[7];

  float* out = (float*)d_out;
  const size_t per = (size_t)BATCH * SEQ * HID;   // 4,194,304 elems
  float* Kc = out + per;                // K_cache output slot (fp32)
  float* Vc = out + 2 * per;            // V_cache output slot (fp32)

  unsigned short* Qbf = (unsigned short*)d_ws;    // bf16 copies for attention
  unsigned short* Kbf = Qbf + per;                // +8MB
  unsigned short* Vbf = Kbf + per;                // +16MB (24MB total ws)

  dim3 g1(HID / NB, (BATCH * SEQ) / MB, 3), b1(256);
  qkv_gemm<<<g1, b1, 0, stream>>>(x, Wq, Wk, Wv, bq, bk, bv,
                                  nullptr, Kc, Vc, Qbf, Kbf, Vbf);

  dim3 g2(SEQ / QB, NH, BATCH), b2(256);
  attn_kernel<<<g2, b2, 0, stream>>>(Qbf, Kbf, Vbf, pad, out);
}
